// TinyGPT_64690797412750
// MI455X (gfx1250) — matrix operand order repaired and checked by reading, not compiled, not run on hardware
//
#include <hip/hip_runtime.h>
#include <hip/hip_bf16.h>
#include <math.h>
#include <stdint.h>

// ---------------- problem constants ----------------
#define CB   2
#define CT   2048
#define CD   1024
#define CH   16
#define CHD  64
#define CL   4
#define CV   32000
#define CDFF 4096
#define CM   (CB * CT)   // 4096 rows of activations

typedef _Float16 v16h __attribute__((ext_vector_type(16)));
typedef _Float16 v8h  __attribute__((ext_vector_type(8)));
typedef float    v8f  __attribute__((ext_vector_type(8)));
typedef unsigned int u32x4 __attribute__((ext_vector_type(4)));
typedef int          i32x8 __attribute__((ext_vector_type(8)));
typedef int          i32x4 __attribute__((ext_vector_type(4)));

// ---------------- WMMA fragment loaders ----------------
// A-matrix 16x32 f16 layout (ISA 7.12.2): lane L holds row M = L&15.
//   lanes 0-15 : elems 0..7 = K 0..7,   elems 8..15 = K 16..23
//   lanes 16-31: elems 0..7 = K 8..15,  elems 8..15 = K 24..31
__device__ __forceinline__ v16h load_a_frag(const _Float16* rowk, int lane) {
    const _Float16* p = rowk + ((lane & 16) ? 8 : 0);
    v8h lo = *(const v8h*)p;
    v8h hi = *(const v8h*)(p + 16);
    return __builtin_shufflevector(lo, hi, 0,1,2,3,4,5,6,7,8,9,10,11,12,13,14,15);
}
// B-matrix 32x16 f16 layout: lane L holds column N = L&15.
//   lanes 0-15 : elems 0..7 = K 0..7,   elems 8..15 = K 16..23
//   lanes 16-31: elems 0..7 = K 8..15,  elems 8..15 = K 24..31
// rowk points at element [col][k0] of a [N][K] ("B transposed") buffer.
__device__ __forceinline__ v16h load_b_frag(const _Float16* rowk, int lane) {
    const _Float16* p = rowk + ((lane & 16) ? 8 : 0);
    v8h lo = *(const v8h*)p;
    v8h hi = *(const v8h*)(p + 16);
    return __builtin_shufflevector(lo, hi, 0,1,2,3,4,5,6,7,8,9,10,11,12,13,14,15);
}

__device__ __forceinline__ float rowmax16(float v) {
    v = fmaxf(v, __shfl_xor(v, 1, 32));
    v = fmaxf(v, __shfl_xor(v, 2, 32));
    v = fmaxf(v, __shfl_xor(v, 4, 32));
    v = fmaxf(v, __shfl_xor(v, 8, 32));
    return v;
}
__device__ __forceinline__ float rowsum16(float v) {
    v += __shfl_xor(v, 1, 32);
    v += __shfl_xor(v, 2, 32);
    v += __shfl_xor(v, 4, 32);
    v += __shfl_xor(v, 8, 32);
    return v;
}

// ---------------- Tensor Data Mover: 2D tile load (32 elems x 128 rows, f16) ----------
// D# per CDNA5 ISA; 6-arg builtin form
// (u32x4 g0, i32x8 g1, i32x4 g2, i32x4 g3, i32x8 extra, i32 cpol).
// group0: [1:0]=count=1, [63:32]=lds_addr, [120:64]=global_addr, [127:126]=type(2)
// group1: [17:16]=data_size(1 => 2B), dim0=32, dim1=128, tile0=32, tile1=128,
//         dim0_stride = ld_elems. groups 2/3 zero (<=2D tensor).
__device__ __forceinline__ void tdm_load_2d(void* lds_dst, const void* gsrc, int ld_elems) {
    unsigned long long ga = (unsigned long long)(uintptr_t)gsrc;
    u32x4 g0 = { 1u,
                 (unsigned)(uintptr_t)lds_dst,
                 (unsigned)(ga & 0xffffffffull),
                 (unsigned)((ga >> 32) & 0x01ffffffull) | 0x80000000u };
    i32x8 g1 = { 0x00010000,        // data_size = 2 bytes
                 (32 << 16),        // tensor_dim0[15:0] = 32 (bits 63:48)
                 (128 << 16),       // tensor_dim1[15:0] = 128 (bits 95:80)
                 (32 << 16),        // tile_dim0 = 32 (bits 127:112)
                 128,               // tile_dim1 = 128 (bits 143:128)
                 ld_elems,          // tensor_dim0_stride[31:0]
                 0, 0 };
    i32x4 gz4 = { 0, 0, 0, 0 };
    i32x8 gz8 = { 0, 0, 0, 0, 0, 0, 0, 0 };
    __builtin_amdgcn_tensor_load_to_lds(g0, g1, gz4, gz4, gz8, 0);
}

// ---------------- small utility kernels ----------------
__global__ void cvt_kernel(const float* __restrict__ in, _Float16* __restrict__ out, int n) {
    int i = blockIdx.x * 256 + threadIdx.x;
    if (i < n) out[i] = (_Float16)in[i];
}

__global__ void embed_kernel(const int* __restrict__ idx, const float* __restrict__ emb,
                             float* __restrict__ x) {
    int i = blockIdx.x * 256 + threadIdx.x;   // CM*CD = 4M
    if (i >= CM * CD) return;
    int d  = i & (CD - 1);
    int bt = i >> 10;
    x[i] = emb[(size_t)idx[bt] * CD + d];
}

// W [K x N] f32  ->  Wt [N x K] f16, LDS-tiled transpose
__global__ __launch_bounds__(256)
void transcvt_kernel(const float* __restrict__ W, _Float16* __restrict__ Wt, int K, int N) {
    __shared__ _Float16 tile[32][33];
    int n0 = blockIdx.x * 32, k0 = blockIdx.y * 32;
    int tx = threadIdx.x & 31, ty = threadIdx.x >> 5;  // ty 0..7
#pragma unroll
    for (int j = 0; j < 32; j += 8)
        tile[ty + j][tx] = (_Float16)W[(size_t)(k0 + ty + j) * N + n0 + tx];
    __syncthreads();
#pragma unroll
    for (int j = 0; j < 32; j += 8)
        Wt[(size_t)(n0 + ty + j) * K + k0 + tx] = tile[tx][ty + j];
}

// LayerNorm: one row (D=1024) per block, emits f16 for the next GEMM's A operand
__global__ __launch_bounds__(256)
void ln_kernel(const float* __restrict__ x, const float* __restrict__ w,
               const float* __restrict__ b, _Float16* __restrict__ out) {
    __shared__ float red[256];
    int row = blockIdx.x;
    const float* xr = x + (size_t)row * CD;
    float s = 0.f;
    for (int d = threadIdx.x; d < CD; d += 256) s += xr[d];
    red[threadIdx.x] = s; __syncthreads();
    for (int st = 128; st > 0; st >>= 1) {
        if (threadIdx.x < st) red[threadIdx.x] += red[threadIdx.x + st];
        __syncthreads();
    }
    float mean = red[0] / CD;
    __syncthreads();
    float vs = 0.f;
    for (int d = threadIdx.x; d < CD; d += 256) { float dv = xr[d] - mean; vs += dv * dv; }
    red[threadIdx.x] = vs; __syncthreads();
    for (int st = 128; st > 0; st >>= 1) {
        if (threadIdx.x < st) red[threadIdx.x] += red[threadIdx.x + st];
        __syncthreads();
    }
    float inv = rsqrtf(red[0] / CD + 1e-5f);
    for (int d = threadIdx.x; d < CD; d += 256)
        out[(size_t)row * CD + d] = (_Float16)((xr[d] - mean) * inv * w[d] + b[d]);
}

// RoPE + head-major relayout from fused QKV [M, 3D].
// Writes q,k as [B,H,T,HD] f16 and v as [B,H,HD,T] f16.
__global__ void rope_prep_kernel(const float* __restrict__ qkv32, _Float16* __restrict__ q16,
                                 _Float16* __restrict__ k16, _Float16* __restrict__ vt16) {
    int tid = blockIdx.x * 256 + threadIdx.x;       // CM * H * HD/2 = 2M
    if (tid >= CM * CH * (CHD / 2)) return;
    int i  = tid & 31;                // pair index within head (HD/2 = 32)
    int h  = (tid >> 5) & (CH - 1);
    int bt = tid >> 9;
    int t  = bt & (CT - 1);
    int bb = bt >> 11;
    float fr  = __expf(-(float)(2 * i) * (9.210340371976184f / (float)CHD)); // 10000^(-2i/HD)
    float ang = (float)t * fr;
    float sn, cs;
    __sincosf(ang, &sn, &cs);
    size_t src = (size_t)bt * (3 * CD) + h * CHD + 2 * i;
    float qe = qkv32[src],          qo = qkv32[src + 1];
    float ke = qkv32[src + CD],     ko = qkv32[src + CD + 1];
    float ve = qkv32[src + 2 * CD], vo = qkv32[src + 2 * CD + 1];
    size_t dst = ((size_t)(bb * CH + h) * CT + t) * CHD + 2 * i;
    q16[dst]     = (_Float16)(qe * cs - qo * sn);
    q16[dst + 1] = (_Float16)(qe * sn + qo * cs);
    k16[dst]     = (_Float16)(ke * cs - ko * sn);
    k16[dst + 1] = (_Float16)(ke * sn + ko * cs);
    size_t vdst = ((size_t)(bb * CH + h) * CHD + 2 * i) * CT + t;
    vt16[vdst]      = (_Float16)ve;
    vt16[vdst + CT] = (_Float16)vo;
}

// ---------------- WMMA GEMM: C[M,N] = epilogue(A[M,K] @ Bt[N,K]^T) ----------------
// 128x128 block tile, 4 waves (2x2), each wave a 64x64 tile (16 f32 WMMA accumulators).
// A/B 128x32 K-slabs staged through LDS by the Tensor Data Mover (wave 0 issues),
// double-buffered, synchronized with s_wait_tensorcnt + workgroup barriers.
template <bool GELU, bool F16OUT>
__global__ __launch_bounds__(128)
void gemm_kernel(const _Float16* __restrict__ A, const _Float16* __restrict__ Bt,
                 const float* __restrict__ bias, const float* res, void* out,
                 int M, int N, int K) {
    __shared__ _Float16 lsA[2][128 * 32];
    __shared__ _Float16 lsB[2][128 * 32];
    int lane = threadIdx.x & 31;
    int wid  = threadIdx.x >> 5;
    int ml   = lane & 15;
    int mbase = blockIdx.y * 128, nbase = blockIdx.x * 128;
    int wm = (wid >> 1) * 64, wn = (wid & 1) * 64;

    if (wid == 0) {   // prologue: fill buffer 0
        tdm_load_2d(&lsA[0][0], A + (size_t)mbase * K, K);
        tdm_load_2d(&lsB[0][0], Bt + (size_t)nbase * K, K);
    }
    v8f acc[4][4] = {};
    int nk = K / 32;
    for (int ki = 0; ki < nk; ++ki) {
        int cur = ki & 1, nxt = cur ^ 1;
        if (wid == 0) {
            if (ki + 1 < nk) {
                tdm_load_2d(&lsA[nxt][0], A + (size_t)mbase * K + (ki + 1) * 32, K);
                tdm_load_2d(&lsB[nxt][0], Bt + (size_t)nbase * K + (ki + 1) * 32, K);
                __builtin_amdgcn_s_wait_tensorcnt(2);  // current pair landed
            } else {
                __builtin_amdgcn_s_wait_tensorcnt(0);
            }
        }
        __syncthreads();
        const _Float16* tA = &lsA[cur][0];
        const _Float16* tB = &lsB[cur][0];
        v16h a[4];
#pragma unroll
        for (int mt = 0; mt < 4; ++mt)
            a[mt] = load_a_frag(tA + (size_t)(wm + mt * 16 + ml) * 32, lane);
#pragma unroll
        for (int nt = 0; nt < 4; ++nt) {
            v16h bfr = load_b_frag(tB + (size_t)(wn + nt * 16 + ml) * 32, lane);
#pragma unroll
            for (int mt = 0; mt < 4; ++mt)
                acc[mt][nt] = __builtin_amdgcn_wmma_f32_16x16x32_f16(
                    false, a[mt], false, bfr, (short)0, acc[mt][nt], false, false);
        }
        __syncthreads();   // all waves done reading 'cur' before it is refilled
    }
    int m0 = mbase + wm, n0 = nbase + wn;
#pragma unroll
    for (int mt = 0; mt < 4; ++mt)
#pragma unroll
        for (int nt = 0; nt < 4; ++nt) {
            int n = n0 + nt * 16 + ml;
            float bv = bias ? bias[n] : 0.f;
#pragma unroll
            for (int e = 0; e < 8; ++e) {
                int m = m0 + mt * 16 + ((lane & 16) ? 8 + e : e);
                size_t off = (size_t)m * N + n;
                float v = acc[mt][nt][e] + bv;
                if (res) v += res[off];
                if (GELU) v = 0.5f * v * (1.f + erff(v * 0.70710678118654752f));
                if (F16OUT) ((_Float16*)out)[off] = (_Float16)v;
                else        ((float*)out)[off] = v;
            }
        }
}

// ---------------- flash attention ----------------
// Grid: x = B*H, y = T/128. 4 waves/block; each wave owns a 32-query tile.
// Streams keys in blocks of 32, online softmax, P staged via LDS (C-layout -> A-layout).
__global__ __launch_bounds__(128)
void flash_kernel(const _Float16* __restrict__ q16, const _Float16* __restrict__ k16,
                  const _Float16* __restrict__ vt16, _Float16* __restrict__ yout) {
    __shared__ _Float16 pls[4][32 * 32];
    const float SCALE = 0.125f;   // 1/sqrt(64)
    int lane = threadIdx.x & 31;
    int wid  = threadIdx.x >> 5;
    int ml   = lane & 15;
    int bh = blockIdx.x;
    int bb = bh >> 4, hh = bh & 15;
    int qb = (blockIdx.y * 4 + wid) * 32;

    const _Float16* qbase = q16  + (size_t)bh * CT * CHD;
    const _Float16* kbase = k16  + (size_t)bh * CT * CHD;
    const _Float16* vbase = vt16 + (size_t)bh * CHD * CT;

    v16h qa[2][2];
#pragma unroll
    for (int mt = 0; mt < 2; ++mt)
#pragma unroll
        for (int kk = 0; kk < 2; ++kk)
            qa[mt][kk] = load_a_frag(qbase + (size_t)(qb + mt * 16 + ml) * CHD + kk * 32, lane);

    v8f y[2][4] = {};
    float mrow[2][8], lrow[2][8];
#pragma unroll
    for (int mt = 0; mt < 2; ++mt)
#pragma unroll
        for (int e = 0; e < 8; ++e) { mrow[mt][e] = -3.0e38f; lrow[mt][e] = 0.f; }

    int nsteps = qb / 32 + 1;
    for (int s = 0; s < nsteps; ++s) {
        int kb = s * 32;
        v8f sc[2][2] = {};
#pragma unroll
        for (int nt = 0; nt < 2; ++nt)
#pragma unroll
            for (int kk = 0; kk < 2; ++kk) {
                v16h bk = load_b_frag(kbase + (size_t)(kb + nt * 16 + ml) * CHD + kk * 32, lane);
#pragma unroll
                for (int mt = 0; mt < 2; ++mt)
                    sc[mt][nt] = __builtin_amdgcn_wmma_f32_16x16x32_f16(
                        false, qa[mt][kk], false, bk, (short)0, sc[mt][nt], false, false);
            }
        // online softmax update, P -> LDS as f16
#pragma unroll
        for (int mt = 0; mt < 2; ++mt) {
#pragma unroll
            for (int e = 0; e < 8; ++e) {
                int roff = (lane & 16) ? 8 + e : e;
                int qr = qb + mt * 16 + roff;
                float v0 = sc[mt][0][e] * SCALE;
                float v1 = sc[mt][1][e] * SCALE;
                if (kb + ml      > qr) v0 = -1e30f;
                if (kb + 16 + ml > qr) v1 = -1e30f;
                float rm = rowmax16(fmaxf(v0, v1));
                float mnew = fmaxf(mrow[mt][e], rm);
                float p0 = __expf(v0 - mnew);
                float p1 = __expf(v1 - mnew);
                float rs = rowsum16(p0 + p1);
                float corr = __expf(mrow[mt][e] - mnew);
                mrow[mt][e] = mnew;
                lrow[mt][e] = lrow[mt][e] * corr + rs;
#pragma unroll
                for (int nt = 0; nt < 4; ++nt) y[mt][nt][e] *= corr;
                int lr = mt * 16 + roff;
                pls[wid][lr * 32 + ml]      = (_Float16)p0;
                pls[wid][lr * 32 + 16 + ml] = (_Float16)p1;
            }
        }
        __syncthreads();
        v16h pa[2];
#pragma unroll
        for (int mt = 0; mt < 2; ++mt)
            pa[mt] = load_a_frag(&pls[wid][(mt * 16 + ml) * 32], lane);
#pragma unroll
        for (int nt = 0; nt < 4; ++nt) {
            v16h vb = load_b_frag(vbase + (size_t)(nt * 16 + ml) * CT + kb, lane);
#pragma unroll
            for (int mt = 0; mt < 2; ++mt)
                y[mt][nt] = __builtin_amdgcn_wmma_f32_16x16x32_f16(
                    false, pa[mt], false, vb, (short)0, y[mt][nt], false, false);
        }
        __syncthreads();
    }
    // normalize and write y back as [B*T, D] f16 (columns = h*HD + hd)
#pragma unroll
    for (int mt = 0; mt < 2; ++mt)
#pragma unroll
        for (int nt = 0; nt < 4; ++nt)
#pragma unroll
            for (int e = 0; e < 8; ++e) {
                int roff = (lane & 16) ? 8 + e : e;
                int qr = qb + mt * 16 + roff;
                size_t off = ((size_t)bb * CT + qr) * CD + hh * CHD + nt * 16 + ml;
                yout[off] = (_Float16)(y[mt][nt][e] / lrow[mt][e]);
            }
}

// ---------------- host orchestration ----------------
extern "C" void kernel_launch(void* const* d_in, const int* in_sizes, int n_in,
                              void* d_out, int out_size, void* d_ws, size_t ws_size,
                              hipStream_t stream) {
    (void)in_sizes; (void)n_in; (void)out_size; (void)ws_size;
    const int*   idx  = (const int*)  d_in[0];
    const float* tok  = (const float*)d_in[1];
    const float* ln1w = (const float*)d_in[2];
    const float* ln1b = (const float*)d_in[3];
    const float* Wq   = (const float*)d_in[4];
    const float* Wk   = (const float*)d_in[5];
    const float* Wv   = (const float*)d_in[6];
    const float* Wp   = (const float*)d_in[7];
    const float* ln2w = (const float*)d_in[8];
    const float* ln2b = (const float*)d_in[9];
    const float* W1   = (const float*)d_in[10];
    const float* b1   = (const float*)d_in[11];
    const float* W2   = (const float*)d_in[12];
    const float* b2   = (const float*)d_in[13];
    const float* lnfw = (const float*)d_in[14];
    const float* lnfb = (const float*)d_in[15];
    float* logits = (float*)d_out;

    char* p = (char*)d_ws;
    auto alloc = [&](size_t bytes) -> void* {
        void* r = (void*)p;
        p += (bytes + 255) & ~(size_t)255;
        return r;
    };
    _Float16* emb16  = (_Float16*)alloc((size_t)CV * CD * 2);
    float*    x      = (float*)   alloc((size_t)CM * CD * 4);
    _Float16* h16    = (_Float16*)alloc((size_t)CM * CD * 2);
    float*    qkv32  = (float*)   alloc((size_t)CM * 3 * CD * 4);
    _Float16* q16    = (_Float16*)alloc((size_t)CM * CD * 2);
    _Float16* k16    = (_Float16*)alloc((size_t)CM * CD * 2);
    _Float16* vt16   = (_Float16*)alloc((size_t)CM * CD * 2);
    _Float16* y16    = (_Float16*)alloc((size_t)CM * CD * 2);
    _Float16* g16    = (_Float16*)alloc((size_t)CM * CDFF * 2);
    _Float16* wqkvt  = (_Float16*)alloc((size_t)3 * CD * CD * 2);  // [3D][D] stacked
    _Float16* wpt    = (_Float16*)alloc((size_t)CD * CD * 2);
    _Float16* w1t    = (_Float16*)alloc((size_t)CDFF * CD * 2);
    _Float16* w2t    = (_Float16*)alloc((size_t)CD * CDFF * 2);

    // token embedding: f16 copy (logits B operand, already [N][K]) + f32 gather
    cvt_kernel<<<(CV * CD + 255) / 256, 256, 0, stream>>>(tok, emb16, CV * CD);
    embed_kernel<<<(CM * CD + 255) / 256, 256, 0, stream>>>(idx, tok, x);

    for (int l = 0; l < CL; ++l) {
        size_t wo = (size_t)l * CD * CD;
        transcvt_kernel<<<dim3(CD / 32, CD / 32), 256, 0, stream>>>(
            Wq + wo, wqkvt, CD, CD);
        transcvt_kernel<<<dim3(CD / 32, CD / 32), 256, 0, stream>>>(
            Wk + wo, wqkvt + (size_t)CD * CD, CD, CD);
        transcvt_kernel<<<dim3(CD / 32, CD / 32), 256, 0, stream>>>(
            Wv + wo, wqkvt + (size_t)2 * CD * CD, CD, CD);
        transcvt_kernel<<<dim3(CD / 32, CD / 32), 256, 0, stream>>>(Wp + wo, wpt, CD, CD);
        transcvt_kernel<<<dim3(CDFF / 32, CD / 32), 256, 0, stream>>>(
            W1 + (size_t)l * CD * CDFF, w1t, CD, CDFF);
        transcvt_kernel<<<dim3(CD / 32, CDFF / 32), 256, 0, stream>>>(
            W2 + (size_t)l * CDFF * CD, w2t, CDFF, CD);

        ln_kernel<<<CM, 256, 0, stream>>>(x, ln1w + l * CD, ln1b + l * CD, h16);

        // fused QKV: [M,3D] = h @ [Wq|Wk|Wv]
        gemm_kernel<false, false><<<dim3(3 * CD / 128, CM / 128), 128, 0, stream>>>(
            h16, wqkvt, nullptr, nullptr, qkv32, CM, 3 * CD, CD);

        rope_prep_kernel<<<(CM * CH * (CHD / 2) + 255) / 256, 256, 0, stream>>>(
            qkv32, q16, k16, vt16);

        flash_kernel<<<dim3(CB * CH, CT / 128), 128, 0, stream>>>(q16, k16, vt16, y16);

        // x = x + y @ Wp
        gemm_kernel<false, false><<<dim3(CD / 128, CM / 128), 128, 0, stream>>>(
            y16, wpt, nullptr, x, x, CM, CD, CD);

        ln_kernel<<<CM, 256, 0, stream>>>(x, ln2w + l * CD, ln2b + l * CD, h16);

        // g = gelu(h @ W1 + b1), f16 output feeds next GEMM directly
        gemm_kernel<true, true><<<dim3(CDFF / 128, CM / 128), 128, 0, stream>>>(
            h16, w1t, b1 + (size_t)l * CDFF, nullptr, g16, CM, CDFF, CD);

        // x = x + g @ W2 + b2
        gemm_kernel<false, false><<<dim3(CD / 128, CM / 128), 128, 0, stream>>>(
            g16, w2t, b2 + (size_t)l * CD, x, x, CM, CD, CDFF);
    }

    ln_kernel<<<CM, 256, 0, stream>>>(x, lnfw, lnfb, h16);

    // logits = lnf(x) @ tok_emb.T  -> Bt layout is tok_emb itself ([V][D] = [N][K])
    gemm_kernel<false, false><<<dim3(CV / 128, CM / 128), 128, 0, stream>>>(
        h16, emb16, nullptr, nullptr, logits, CM, CV, CD);
}
